// RGCNModel_71365176590645
// MI455X (gfx1250) — matrix operand order repaired and checked by reading, not compiled, not run on hardware
//
#include <hip/hip_runtime.h>

// ---------------------------------------------------------------------------
// RGCN forward on gfx1250 (CDNA5): WMMA bf16 GEMMs + edge scatter.
// GEMM: BM=128 x BN=64 x BK=32 tiles, 4 waves, 8 WMMA/wave/k-step,
// double-buffered LDS, software-pipelined staging, async global->LDS B tile,
// single-instruction bf16 packing (v_perm_b32) in the hot conversion paths.
// ---------------------------------------------------------------------------

typedef __attribute__((ext_vector_type(16))) __bf16 v16bf;
typedef __attribute__((ext_vector_type(2)))  __bf16 v2bf;
typedef __attribute__((ext_vector_type(8)))  float  v8f;
typedef int v4i_t __attribute__((vector_size(16)));           // int __vector(4)
typedef __attribute__((address_space(1))) v4i_t* gptr_v4i;    // global
typedef __attribute__((address_space(3))) v4i_t* lptr_v4i;    // LDS

union FragU { unsigned u[8]; v16bf v; };

#if __has_builtin(__builtin_amdgcn_global_load_async_to_lds_b128)
#define ASYNC_LDS 1
#else
#define ASYNC_LDS 0
#endif

static __device__ __forceinline__ void wait_asynccnt0() {
#if __has_builtin(__builtin_amdgcn_s_wait_asynccnt)
    __builtin_amdgcn_s_wait_asynccnt(0);
#else
    asm volatile("s_wait_asynccnt 0x0" ::: "memory");
#endif
}

// exact RNE scalar convert (used for once-per-launch weight conversion)
static __device__ __forceinline__ unsigned short f2bf(float f) {
    union { float f; unsigned u; } v; v.f = f;
    unsigned r = v.u + 0x7FFFu + ((v.u >> 16) & 1u);
    return (unsigned short)(r >> 16);
}
static __device__ __forceinline__ float bf2f(unsigned short b) {
    union { unsigned u; float f; } v; v.u = ((unsigned)b) << 16;
    return v.f;
}
// hot-path pack: two floats -> bf16x2 in ONE VALU op.
// v_cvt_pk_bf16_f32 if the toolchain has it, else v_perm_b32 (truncate).
static __device__ __forceinline__ unsigned pack2(float a, float b) {
#if __has_builtin(__builtin_amdgcn_cvt_pk_bf16_f32)
    union { v2bf v; unsigned u; } c;
    c.v = __builtin_amdgcn_cvt_pk_bf16_f32(a, b);
    return c.u;
#else
    union { float f; unsigned u; } ua, ub; ua.f = a; ub.f = b;
    // out bytes (LSB..MSB) = {a.b2, a.b3, b.b2, b.b3}; S0=b (idx 4-7), S1=a (idx 0-3)
    return __builtin_amdgcn_perm(ub.u, ua.u, 0x07060302u);
#endif
}

// --------------------------- weight conversion -----------------------------

// dst[n*K+k] = bf16(src[k*N+n])   (transpose fp32 [K,N] -> bf16 [N,K])
__global__ void k_transpose_bf16(const float* __restrict__ src,
                                 unsigned short* __restrict__ dst, int K, int N) {
    int k = blockIdx.x * blockDim.x + threadIdx.x;
    int n = blockIdx.y;
    if (k < K) dst[(size_t)n * K + k] = f2bf(src[(size_t)k * N + n]);
}

// bcatT[n][k] (1024 x 256): row n<256 = root col n, else W[r] col (n&255).
// Also biasv[n] = [conv_bias | 0 | 0 | 0].
__global__ void k_build_bcat_t(const float* __restrict__ root,
                               const float* __restrict__ W,
                               const float* __restrict__ bias,
                               unsigned short* __restrict__ bcatT,
                               float* __restrict__ biasv) {
    int k = threadIdx.x;            // 256 threads
    int n = blockIdx.y;             // 0..1023
    float v;
    if (n < 256) v = root[k * 256 + n];
    else         v = W[(size_t)((n >> 8) - 1) * 65536 + k * 256 + (n & 255)];
    bcatT[(size_t)n * 256 + k] = f2bf(v);
    if (k == 0) biasv[n] = (n < 256) ? bias[n] : 0.0f;
}

// ------------------------------ WMMA GEMM ----------------------------------
// C[M,N] = op(A[M,K] @ Bt[N,K]^T + bias[N]); A fp32-or-bf16 row-major,
// Bt bf16 [N][K], C bf16. 128 threads (4 waves); wave w owns rows
// [w*32, w*32+32): 2 A-frags x 4 B-frags -> 8 WMMA per k-step.
// Out-of-range A rows are address-clamped but NOT zeroed: they only feed
// accumulator rows >= M, which the guarded epilogue never stores.

template <bool A_FP32, bool RELU>
__global__ __launch_bounds__(128) void
k_gemm(const void* __restrict__ Av, const unsigned short* __restrict__ Bt,
       const float* __restrict__ bias, unsigned short* __restrict__ C,
       int M, int N, int K) {
    constexpr int LDA = 20;                     // uints/row: 16 data + 4 pad (80B)
    __shared__ unsigned ldsA[2][128 * LDA];     // A tiles 128(m) x 32(k) bf16
    __shared__ unsigned ldsB[2][64 * LDA];      // B tiles  64(n) x 32(k) bf16

    const int m0   = blockIdx.y * 128;
    const int n0   = blockIdx.x * 64;
    const int tid  = threadIdx.x;
    const int wave = tid >> 5;
    const int lane = tid & 31;
    const int hl   = lane >> 4;
    const int l16  = lane & 15;

    v8f acc[2][4];
    #pragma unroll
    for (int s = 0; s < 2; ++s)
        #pragma unroll
        for (int t = 0; t < 4; ++t)
            #pragma unroll
            for (int j = 0; j < 8; ++j) acc[s][t][j] = 0.0f;

    float4 raf[8];          // staged A (fp32 path)
    uint4  rah[4];          // staged A (bf16 path)
    uint4  rb[2];           // staged B (fallback path)

    auto loadA = [&](int k0) {
        if (A_FP32) {
            const float* A = (const float*)Av;
            #pragma unroll
            for (int i = 0; i < 8; ++i) {
                int idx = tid + i * 128;                 // float4 idx 0..1023
                int row = idx >> 3, col = (idx & 7) * 4;
                int gm  = m0 + row;
                raf[i] = *(const float4*)(A + (size_t)(gm < M ? gm : 0) * K + k0 + col);
            }
        } else {
            const unsigned short* A = (const unsigned short*)Av;
            #pragma unroll
            for (int i = 0; i < 4; ++i) {
                int idx = tid + i * 128;                 // uint4 idx 0..511
                int row = idx >> 2, col8 = (idx & 3) * 8;
                int gm  = m0 + row;
                rah[i] = *(const uint4*)(A + (size_t)(gm < M ? gm : 0) * K + k0 + col8);
            }
        }
    };
    auto storeA = [&](unsigned* LA) {
        if (A_FP32) {
            #pragma unroll
            for (int i = 0; i < 8; ++i) {
                int idx = tid + i * 128;
                int row = idx >> 3, col = (idx & 7) * 4;
                uint2 u; u.x = pack2(raf[i].x, raf[i].y); u.y = pack2(raf[i].z, raf[i].w);
                *(uint2*)&LA[row * LDA + (col >> 1)] = u;
            }
        } else {
            #pragma unroll
            for (int i = 0; i < 4; ++i) {
                int idx = tid + i * 128;
                int row = idx >> 2, col8 = (idx & 3) * 8;
                *(uint4*)&LA[row * LDA + (col8 >> 1)] = rah[i];
            }
        }
    };
#if ASYNC_LDS
    auto stageB_async = [&](int k0, unsigned* LB) {
        #pragma unroll
        for (int i = 0; i < 2; ++i) {
            int idx = tid + i * 128;                     // uint4 idx 0..255
            int row = idx >> 2, col8 = (idx & 3) * 8;
            const unsigned short* g = Bt + (size_t)(n0 + row) * K + k0 + col8;
            __builtin_amdgcn_global_load_async_to_lds_b128(
                (gptr_v4i)g, (lptr_v4i)&LB[row * LDA + (col8 >> 1)], 0, 0);
        }
    };
#else
    auto loadB = [&](int k0) {
        #pragma unroll
        for (int i = 0; i < 2; ++i) {
            int idx = tid + i * 128;
            int row = idx >> 2, col8 = (idx & 3) * 8;
            rb[i] = *(const uint4*)(Bt + (size_t)(n0 + row) * K + k0 + col8);
        }
    };
    auto storeB = [&](unsigned* LB) {
        #pragma unroll
        for (int i = 0; i < 2; ++i) {
            int idx = tid + i * 128;
            int row = idx >> 2, col8 = (idx & 3) * 8;
            *(uint4*)&LB[row * LDA + (col8 >> 1)] = rb[i];
        }
    };
#endif

    // ---- prologue: stage tile k0=0 into buffer 0 ----
    loadA(0);
#if ASYNC_LDS
    stageB_async(0, ldsB[0]);
#else
    loadB(0);
#endif
    storeA(ldsA[0]);
#if ASYNC_LDS
    wait_asynccnt0();
#else
    storeB(ldsB[0]);
#endif
    __syncthreads();

    int stage = 0;
    for (int k0 = 0; k0 < K; k0 += 32) {
        const int  nxt  = stage ^ 1;
        const bool more = (k0 + 32) < K;

        // issue next tile's global traffic before computing (overlap)
        if (more) {
            loadA(k0 + 32);
#if ASYNC_LDS
            stageB_async(k0 + 32, ldsB[nxt]);
#else
            loadB(k0 + 32);
#endif
        }

        // ---- preload all fragments, then 8 WMMA ----
        const unsigned* LA = ldsA[stage];
        const unsigned* LB = ldsB[stage];
        FragU fa0, fa1, fb[4];
        const int ar0 = wave * 32 + l16;
        #pragma unroll
        for (int j = 0; j < 8; ++j) {
            int ui = hl * 8 + j;   // koff = 2*ui
            fa0.u[j] = LA[ar0 * LDA + ui];
            fa1.u[j] = LA[(ar0 + 16) * LDA + ui];
        }
        #pragma unroll
        for (int t = 0; t < 4; ++t)
            #pragma unroll
            for (int j = 0; j < 8; ++j)
                fb[t].u[j] = LB[(t * 16 + l16) * LDA + hl * 8 + j];  // koff = hl*16+2j
        #pragma unroll
        for (int t = 0; t < 4; ++t) {
            acc[0][t] = __builtin_amdgcn_wmma_f32_16x16x32_bf16(
                false, fa0.v, false, fb[t].v, (short)0, acc[0][t], false, false);
            acc[1][t] = __builtin_amdgcn_wmma_f32_16x16x32_bf16(
                false, fa1.v, false, fb[t].v, (short)0, acc[1][t], false, false);
        }

        // ---- commit next tile to the other LDS buffer ----
        if (more) {
            storeA(ldsA[nxt]);
#if ASYNC_LDS
            wait_asynccnt0();
#else
            storeB(ldsB[nxt]);
#endif
        }
        __syncthreads();
        stage = nxt;
    }

    // ---- epilogue: packed cvt, halves stored via b16 / d16_hi stores ----
    #pragma unroll
    for (int s = 0; s < 2; ++s)
        #pragma unroll
        for (int t = 0; t < 4; ++t) {
            const int gcol = n0 + t * 16 + l16;
            const float bb = bias[gcol];
            #pragma unroll
            for (int r = 0; r < 8; r += 2) {
                int g0 = m0 + wave * 32 + s * 16 + hl * 8 + r;
                float v0 = acc[s][t][r] + bb;
                float v1 = acc[s][t][r + 1] + bb;
                if (RELU) { v0 = fmaxf(v0, 0.0f); v1 = fmaxf(v1, 0.0f); }
                unsigned p = pack2(v0, v1);
                if (g0 < M)     C[(size_t)g0 * N + gcol]       = (unsigned short)(p & 0xFFFFu);
                if (g0 + 1 < M) C[(size_t)(g0 + 1) * N + gcol] = (unsigned short)(p >> 16);
            }
        }
}

// ----------------------------- edge kernels --------------------------------

__global__ void k_zero(float* __restrict__ p, int n) {
    int i = blockIdx.x * blockDim.x + threadIdx.x;
    if (i < n) p[i] = 0.0f;
}

__global__ void k_count(const int* __restrict__ dst, const int* __restrict__ et,
                        float* __restrict__ cnt, int E) {
    int e = blockIdx.x * blockDim.x + threadIdx.x;
    if (e < E) atomicAdd(cnt + (size_t)dst[e] * 3 + et[e], 1.0f);
}

__global__ void k_inv(float* __restrict__ cnt, int n) {
    int i = blockIdx.x * blockDim.x + threadIdx.x;
    if (i < n) cnt[i] = 1.0f / fmaxf(cnt[i], 1.0f);
}

// out[n][c..c+3] = float(t[n][c..c+3])  (root term; bias folded in by GEMM)
__global__ void k_init_out(const unsigned short* __restrict__ t,
                           float* __restrict__ out, int nQuads) {
    int i = blockIdx.x * blockDim.x + threadIdx.x;
    if (i >= nQuads) return;
    int n = i >> 6, c = (i & 63) * 4;
    uint2 u = *(const uint2*)(t + (size_t)n * 1024 + c);
    float4 f;
    f.x = bf2f((unsigned short)(u.x & 0xFFFFu));
    f.y = bf2f((unsigned short)(u.x >> 16));
    f.z = bf2f((unsigned short)(u.y & 0xFFFFu));
    f.w = bf2f((unsigned short)(u.y >> 16));
    ((float4*)out)[i] = f;
}

// 32 threads/edge, 8 cols each: out[dst] += inv[dst,r] * t[src, rel slice]
__global__ void k_scatter(const unsigned short* __restrict__ t,
                          const int* __restrict__ src, const int* __restrict__ dst,
                          const int* __restrict__ et, const float* __restrict__ inv,
                          float* __restrict__ out, int E) {
    int tid = blockIdx.x * blockDim.x + threadIdx.x;
    if (tid >= E * 32) return;
    int e = tid >> 5, q = tid & 31;
    int s = src[e], d = dst[e], r = et[e];
    float sc = inv[(size_t)d * 3 + r];
    uint4 v = *(const uint4*)(t + ((size_t)s * 1024 + 256 + r * 256 + q * 8));
    float* o = out + (size_t)d * 256 + q * 8;
    unsigned u[4] = {v.x, v.y, v.z, v.w};
    #pragma unroll
    for (int j = 0; j < 4; ++j) {
        atomicAdd(o + 2 * j,     bf2f((unsigned short)(u[j] & 0xFFFFu)) * sc);
        atomicAdd(o + 2 * j + 1, bf2f((unsigned short)(u[j] >> 16)) * sc);
    }
}

// pack fp32 out -> bf16 h, four at a time
__global__ void k_out_to_h(const float* __restrict__ out,
                           uint2* __restrict__ h, int nQuads) {
    int i = blockIdx.x * blockDim.x + threadIdx.x;
    if (i >= nQuads) return;
    float4 f = ((const float4*)out)[i];
    uint2 u; u.x = pack2(f.x, f.y); u.y = pack2(f.z, f.w);
    h[i] = u;
}

// logits = hmid(bf16)[rows,128] @ Wc[128,2] + bc[2]
__global__ void k_classifier(const unsigned short* __restrict__ hmid,
                             const float* __restrict__ Wc,
                             const float* __restrict__ bc,
                             float* __restrict__ out, int rows) {
    int tid = blockIdx.x * blockDim.x + threadIdx.x;
    if (tid >= rows * 2) return;
    int row = tid >> 1, c = tid & 1;
    float s = bc[c];
    #pragma unroll 8
    for (int j = 0; j < 128; ++j)
        s += bf2f(hmid[(size_t)row * 128 + j]) * Wc[j * 2 + c];
    out[tid] = s;
}

// ------------------------------- launcher ----------------------------------

extern "C" void kernel_launch(void* const* d_in, const int* in_sizes, int n_in,
                              void* d_out, int out_size, void* d_ws, size_t ws_size,
                              hipStream_t stream) {
    const float* x      = (const float*)d_in[0];
    const int*   eidx   = (const int*)  d_in[1];
    const int*   et     = (const int*)  d_in[2];
    const float* W1     = (const float*)d_in[4];
    const float* b1     = (const float*)d_in[5];
    const float* c1W    = (const float*)d_in[6];
    const float* c1root = (const float*)d_in[7];
    const float* c1bias = (const float*)d_in[8];
    const float* c2W    = (const float*)d_in[9];
    const float* c2root = (const float*)d_in[10];
    const float* c2bias = (const float*)d_in[11];
    const float* W2     = (const float*)d_in[12];
    const float* b2     = (const float*)d_in[13];
    const float* Wc     = (const float*)d_in[14];
    const float* bc     = (const float*)d_in[15];

    const int NN    = in_sizes[0] / 768;     // 100000 nodes
    const int E     = in_sizes[2];           // 400000 edges
    const int batch = out_size / 2;          // 1024
    const int* srcp = eidx;
    const int* dstp = eidx + E;

    char* ws = (char*)d_ws;
    size_t off = 0;
    auto alloc = [&](size_t bytes) -> void* {
        void* p = ws + off;
        off += (bytes + 255) & ~(size_t)255;
        return p;
    };
    unsigned short* h      = (unsigned short*)alloc((size_t)NN * 256 * 2);
    unsigned short* t      = (unsigned short*)alloc((size_t)NN * 1024 * 2);
    float*          outb   = (float*)         alloc((size_t)NN * 256 * 4);
    float*          cnt    = (float*)         alloc((size_t)NN * 3 * 4);
    unsigned short* w1bt   = (unsigned short*)alloc((size_t)256 * 768 * 2);
    unsigned short* bcat1t = (unsigned short*)alloc((size_t)1024 * 256 * 2);
    unsigned short* bcat2t = (unsigned short*)alloc((size_t)1024 * 256 * 2);
    unsigned short* w2bt   = (unsigned short*)alloc((size_t)128 * 256 * 2);
    float*          bv1    = (float*)         alloc(1024 * 4);
    float*          bv2    = (float*)         alloc(1024 * 4);
    unsigned short* hmid   = (unsigned short*)alloc((size_t)batch * 128 * 2);

    const int TB = 256;
    // --- weight conversions (transposed to [N][K] bf16) ---
    k_transpose_bf16<<<dim3(3, 256), 256, 0, stream>>>(W1, w1bt, 768, 256);
    k_transpose_bf16<<<dim3(1, 128), 256, 0, stream>>>(W2, w2bt, 256, 128);
    k_build_bcat_t<<<dim3(1, 1024), 256, 0, stream>>>(c1root, c1W, c1bias, bcat1t, bv1);
    k_build_bcat_t<<<dim3(1, 1024), 256, 0, stream>>>(c2root, c2W, c2bias, bcat2t, bv2);

    const int MBLK = (NN + 127) / 128;
    // --- linear1: h = relu(x @ W1 + b1), fp32 A ---
    k_gemm<true, true><<<dim3(256 / 64, MBLK), 128, 0, stream>>>(
        (const void*)x, w1bt, b1, h, NN, 256, 768);

    // --- two RGCN convs ---
    for (int pass = 0; pass < 2; ++pass) {
        const unsigned short* bcatT = pass ? bcat2t : bcat1t;
        const float* bv = pass ? bv2 : bv1;
        k_gemm<false, false><<<dim3(1024 / 64, MBLK), 128, 0, stream>>>(
            (const void*)h, bcatT, bv, t, NN, 1024, 256);
        k_zero<<<(NN * 3 + TB - 1) / TB, TB, 0, stream>>>(cnt, NN * 3);
        k_count<<<(E + TB - 1) / TB, TB, 0, stream>>>(dstp, et, cnt, E);
        k_inv<<<(NN * 3 + TB - 1) / TB, TB, 0, stream>>>(cnt, NN * 3);
        k_init_out<<<(NN * 64 + TB - 1) / TB, TB, 0, stream>>>(t, outb, NN * 64);
        k_scatter<<<(E * 32 + TB - 1) / TB, TB, 0, stream>>>(t, srcp, dstp, et, cnt, outb, E);
        k_out_to_h<<<(NN * 64 + TB - 1) / TB, TB, 0, stream>>>(outb, (uint2*)h, NN * 64);
    }

    // --- linear2 on first `batch` rows ---
    k_gemm<false, true><<<dim3(128 / 64, (batch + 127) / 128), 128, 0, stream>>>(
        (const void*)h, w2bt, b2, hmid, batch, 128, 256);
    // --- classifier ---
    k_classifier<<<(batch * 2 + TB - 1) / TB, TB, 0, stream>>>(
        hmid, Wc, bc, (float*)d_out, batch);
}
